// MultiHeadAttention_40913858462063
// MI455X (gfx1250) — matrix operand order repaired and checked by reading, not compiled, not run on hardware
//
#include <hip/hip_runtime.h>
#include <hip/hip_bf16.h>

typedef __attribute__((ext_vector_type(16))) __bf16 v16bf;
typedef __attribute__((ext_vector_type(8)))  __bf16 v8bf;
typedef __attribute__((ext_vector_type(8)))  float  v8f;

constexpr int SEQ  = 4096;
constexpr int DM   = 768;
constexpr int NH   = 12;
constexpr int HD   = 64;
constexpr int MT   = SEQ / 16;   // 256 16-row tiles (attention)
constexpr int MT32 = SEQ / 32;   // 128 32-row tiles (GEMMs)
constexpr int NT   = DM / 64;    // 12 N-strips of 64

__device__ __forceinline__ v16bf cat8(v8bf lo, v8bf hi) {
  return __builtin_shufflevector(lo, hi, 0,1,2,3,4,5,6,7,8,9,10,11,12,13,14,15);
}

__device__ __forceinline__ v8f wmma_bf16(v16bf a, v16bf b, v8f c) {
  // D = A(16x32 bf16) * B(32x16 bf16) + C(16x16 f32)
  return __builtin_amdgcn_wmma_f32_16x16x32_bf16(false, a, false, b, (short)0, c,
                                                 false, false);
}

// A-operand: lane holds row l16; 16 consecutive K values at akb+0..15
__device__ __forceinline__ v16bf load_a16(const __bf16* rowp, int k0, int akb) {
  return cat8(*(const v8bf*)(rowp + k0 + akb),
              *(const v8bf*)(rowp + k0 + akb + 8));
}
// B-operand: lane holds col l16; 16 consecutive K values starting at p
__device__ __forceinline__ v16bf load_b16(const __bf16* p) {
  return cat8(*(const v8bf*)p, *(const v8bf*)(p + 8));
}

// ---------------------------------------------------------------------------
// prep: x fp32 -> bf16 row-major; W[K][N] fp32 -> WT[N][K] bf16 (B^T layout)
// ---------------------------------------------------------------------------
__global__ void prep_kernel(const float* __restrict__ x,
                            const float* __restrict__ wq, const float* __restrict__ wk,
                            const float* __restrict__ wv, const float* __restrict__ wo,
                            __bf16* __restrict__ xb,
                            __bf16* __restrict__ wqT, __bf16* __restrict__ wkT,
                            __bf16* __restrict__ wvT, __bf16* __restrict__ woT) {
  int tid = blockIdx.x * blockDim.x + threadIdx.x;
  if (tid < SEQ * DM) xb[tid] = (__bf16)x[tid];
  if (tid < DM * DM) {
    int k = tid / DM, n = tid % DM;
    int t = n * DM + k;
    wqT[t] = (__bf16)wq[tid];
    wkT[t] = (__bf16)wk[tid];
    wvT[t] = (__bf16)wv[tid];
    woT[t] = (__bf16)wo[tid];
  }
}

// ---------------------------------------------------------------------------
// One wave computes a 32x64 f32 tile: C = A[32xK] * BT[64xK]^T
// Software-pipelined: k-step (k0+32) operands load while k0's 8 WMMAs issue.
// unroll 2 lets SSA renaming absorb the double-buffer rotation (no v_mov
// copies each step) while keeping <=3 operand sets live (no spills).
// acc[0..3] = rows 0..15, acc[4..7] = rows 16..31 (4 col-tiles each).
// ---------------------------------------------------------------------------
template <int LDA, int LDB, int K>
__device__ __forceinline__ void gemm_tile_32x64(const __bf16* __restrict__ A, int arow0,
                                                const __bf16* __restrict__ BT, int bcol0,
                                                v8f acc[8]) {
  const int lane = threadIdx.x & 31;
  const int half = lane >> 4;
  const int l16  = lane & 15;
  const int akb  = half * 16;   // A-operand K base per lane-half
  const int bkb  = half * 16;   // B-operand K base per lane-half
  const __bf16* ar = A + (size_t)(arow0 + l16) * LDA;        // rows l16, l16+16
  const __bf16* br = BT + (size_t)(bcol0 + l16) * LDB + bkb; // cols j*16 via imm

  v16bf a0 = load_a16(ar, 0, akb);
  v16bf a1 = load_a16(ar + 16 * LDA, 0, akb);
  v16bf b[4];
#pragma unroll
  for (int j = 0; j < 4; ++j) b[j] = load_b16(br + j * 16 * LDB);

#pragma unroll 2
  for (int k0 = 32; k0 < K; k0 += 32) {
    // preload next k-step into fresh registers (overlaps the WMMAs below)
    v16bf a0n = load_a16(ar, k0, akb);
    v16bf a1n = load_a16(ar + 16 * LDA, k0, akb);
    v16bf bn[4];
#pragma unroll
    for (int j = 0; j < 4; ++j) bn[j] = load_b16(br + j * 16 * LDB + k0);
#pragma unroll
    for (int j = 0; j < 4; ++j) acc[j]     = wmma_bf16(a0, b[j], acc[j]);
#pragma unroll
    for (int j = 0; j < 4; ++j) acc[4 + j] = wmma_bf16(a1, b[j], acc[4 + j]);
    a0 = a0n; a1 = a1n;
#pragma unroll
    for (int j = 0; j < 4; ++j) b[j] = bn[j];
  }
#pragma unroll
  for (int j = 0; j < 4; ++j) acc[j]     = wmma_bf16(a0, b[j], acc[j]);
#pragma unroll
  for (int j = 0; j < 4; ++j) acc[4 + j] = wmma_bf16(a1, b[j], acc[4 + j]);
}

// ---------------------------------------------------------------------------
// QKV projection: Q = x Wq, K = x Wk (row-major bf16), V = x Wv written
// transposed as VT[d_global][s] so it serves directly as a B-operand later.
// ---------------------------------------------------------------------------
__global__ void __launch_bounds__(128, 1)
qkv_kernel(const __bf16* __restrict__ xb,
           const __bf16* __restrict__ wqT, const __bf16* __restrict__ wkT,
           const __bf16* __restrict__ wvT,
           __bf16* __restrict__ Qb, __bf16* __restrict__ Kb,
           __bf16* __restrict__ VTb) {
  int wave = (blockIdx.x * blockDim.x + threadIdx.x) >> 5;
  int lane = threadIdx.x & 31;
  int mat  = wave / (MT32 * NT);
  int rem  = wave % (MT32 * NT);
  int mt   = rem / NT, nt = rem % NT;
  const __bf16* BT = (mat == 0) ? wqT : (mat == 1) ? wkT : wvT;

  v8f acc[8] = {};
  gemm_tile_32x64<DM, DM, DM>(xb, mt * 32, BT, nt * 64, acc);

  int half = lane >> 4, l16 = lane & 15;
  if (mat < 2) {
    __bf16* out = (mat == 0) ? Qb : Kb;
#pragma unroll
    for (int g = 0; g < 2; ++g)
#pragma unroll
      for (int j = 0; j < 4; ++j)
#pragma unroll
        for (int r = 0; r < 8; ++r) {
          int row = mt * 32 + g * 16 + half * 8 + r;
          int col = nt * 64 + j * 16 + l16;
          out[(size_t)row * DM + col] = (__bf16)acc[g * 4 + j][r];
        }
  } else {
#pragma unroll
    for (int g = 0; g < 2; ++g)
#pragma unroll
      for (int j = 0; j < 4; ++j)
#pragma unroll
        for (int r = 0; r < 8; ++r) {
          int row = mt * 32 + g * 16 + half * 8 + r;
          int col = nt * 64 + j * 16 + l16;
          VTb[(size_t)col * SEQ + row] = (__bf16)acc[g * 4 + j][r];  // transposed
        }
  }
}

// ---------------------------------------------------------------------------
// Flash attention: one wave per (head, 16-query tile); key tiles of 32.
// K-operands for tile j0+32 and V-operands for j0 load while softmax runs.
// ---------------------------------------------------------------------------
__global__ void __launch_bounds__(128, 1)
attn_kernel(const __bf16* __restrict__ Qb, const __bf16* __restrict__ Kb,
            const __bf16* __restrict__ VTb, __bf16* __restrict__ Ctxb) {
  __shared__ __bf16 Pbuf[4][16 * 32];  // per-wave P staging (C-layout -> A-layout)

  int wid  = threadIdx.x >> 5;
  int lane = threadIdx.x & 31;
  int wave = blockIdx.x * (blockDim.x >> 5) + wid;
  int h    = wave / MT;      // head
  int qt   = wave % MT;      // query tile
  int q0   = qt * 16;
  int half = lane >> 4, l16 = lane & 15;
  int akb  = half * 16;

  // Q A-operands for the whole key loop: 16x64 = two 16x32 operands
  v16bf qa[2];
  const __bf16* qrow = Qb + (size_t)(q0 + l16) * DM + h * HD;
#pragma unroll
  for (int ks = 0; ks < 2; ++ks) qa[ks] = load_a16(qrow, ks * 32, akb);

  // base pointers for K (score B-operand) and V^T (PV B-operand)
  const __bf16* kbase = Kb + (size_t)l16 * DM + h * HD + half * 16;      // + t*16*DM imm
  const __bf16* vbase = VTb + (size_t)(h * HD + l16) * SEQ + half * 16;  // + j*16*SEQ imm

  v8f o[4] = {};
  float mi[8], li[8];
#pragma unroll
  for (int r = 0; r < 8; ++r) { mi[r] = -1e30f; li[r] = 0.0f; }
  const float scale = 0.125f;  // 1/sqrt(64)

  int jend = q0 + 16;  // causal: keys < q0+16 needed

  // preload first key tile's B-operands: kb[t*2+ks] = K-chunk (tile t, k-step ks)
  v16bf kb[4];
#pragma unroll
  for (int t = 0; t < 2; ++t)
#pragma unroll
    for (int ks = 0; ks < 2; ++ks)
      kb[t * 2 + ks] = load_b16(kbase + t * 16 * DM + ks * 32);

#pragma unroll 1
  for (int j0 = 0; j0 < jend; j0 += 32) {
    // ---- scores: two 16x16 C tiles (keys j0..j0+15, j0+16..j0+31) ----
    v8f s[2] = {};
#pragma unroll
    for (int t = 0; t < 2; ++t)
#pragma unroll
      for (int ks = 0; ks < 2; ++ks)
        s[t] = wmma_bf16(qa[ks], kb[t * 2 + ks], s[t]);

    // ---- issue V loads for this tile and K loads for the next tile now,
    //      so they overlap the whole softmax VALU section ----
    v16bf vb[4];
#pragma unroll
    for (int j = 0; j < 4; ++j) vb[j] = load_b16(vbase + j * 16 * SEQ + j0);

    bool more = (j0 + 32 < jend);
    v16bf kbn[4];
    if (more) {
#pragma unroll
      for (int t = 0; t < 2; ++t)
#pragma unroll
        for (int ks = 0; ks < 2; ++ks)
          kbn[t * 2 + ks] =
              load_b16(kbase + (size_t)(j0 + 32) * DM + t * 16 * DM + ks * 32);
      __builtin_prefetch(Kb + (size_t)(j0 + 64 + l16) * DM + h * HD, 0, 0);
    } else {
#pragma unroll
      for (int i = 0; i < 4; ++i) kbn[i] = kb[i];
    }

    // ---- scale + causal mask (VALU selects; no EXEC divergence) ----
#pragma unroll
    for (int t = 0; t < 2; ++t) {
      int kcol = j0 + t * 16 + l16;
#pragma unroll
      for (int r = 0; r < 8; ++r) {
        int qrow_i = q0 + half * 8 + r;
        float v = s[t][r] * scale;
        s[t][r] = (kcol > qrow_i) ? -1e30f : v;
      }
    }

    // ---- online softmax (row stats replicated per 16-lane half) ----
#pragma unroll
    for (int r = 0; r < 8; ++r) {
      float mx = fmaxf(s[0][r], s[1][r]);
#pragma unroll
      for (int off = 1; off < 16; off <<= 1)
        mx = fmaxf(mx, __shfl_xor(mx, off, 16));
      float mnew = fmaxf(mi[r], mx);
      float alpha = __expf(mi[r] - mnew);
      float p0 = __expf(s[0][r] - mnew);
      float p1 = __expf(s[1][r] - mnew);
      s[0][r] = p0; s[1][r] = p1;
      float rs = p0 + p1;
#pragma unroll
      for (int off = 1; off < 16; off <<= 1)
        rs += __shfl_xor(rs, off, 16);
      li[r] = li[r] * alpha + rs;
      mi[r] = mnew;
#pragma unroll
      for (int j = 0; j < 4; ++j) o[j][r] *= alpha;
    }

    // ---- reshape P (C-layout) -> A-operand via per-wave LDS ----
    __bf16* P = &Pbuf[wid][0];
#pragma unroll
    for (int t = 0; t < 2; ++t)
#pragma unroll
      for (int r = 0; r < 8; ++r)
        P[(half * 8 + r) * 32 + t * 16 + l16] = (__bf16)s[t][r];
    // same-wave LDS RAW: DS ops are in-order per wave; compiler inserts dscnt wait
    const __bf16* prow = P + l16 * 32 + akb;
    v16bf pa = cat8(*(const v8bf*)(prow), *(const v8bf*)(prow + 8));

    // ---- O += P(16x32) * V(32x64): 4 WMMAs, B from transposed V ----
#pragma unroll
    for (int j = 0; j < 4; ++j) o[j] = wmma_bf16(pa, vb[j], o[j]);

#pragma unroll
    for (int i = 0; i < 4; ++i) kb[i] = kbn[i];
  }

  // ---- epilogue: normalize and store ctx (bf16, row-major) ----
#pragma unroll
  for (int r = 0; r < 8; ++r) {
    float inv = 1.0f / li[r];
    int row = q0 + half * 8 + r;
#pragma unroll
    for (int j = 0; j < 4; ++j)
      Ctxb[(size_t)row * DM + h * HD + j * 16 + l16] = (__bf16)(o[j][r] * inv);
  }
}

// ---------------------------------------------------------------------------
// Output projection: out = ctx @ W_out + b_out  (f32 store)
// ---------------------------------------------------------------------------
__global__ void __launch_bounds__(128, 1)
outproj_kernel(const __bf16* __restrict__ Ctxb, const __bf16* __restrict__ woT,
               const float* __restrict__ bias, float* __restrict__ out) {
  int wave = (blockIdx.x * blockDim.x + threadIdx.x) >> 5;
  int lane = threadIdx.x & 31;
  int mt = wave / NT, nt = wave % NT;

  v8f acc[8] = {};
  gemm_tile_32x64<DM, DM, DM>(Ctxb, mt * 32, woT, nt * 64, acc);

  int half = lane >> 4, l16 = lane & 15;
#pragma unroll
  for (int j = 0; j < 4; ++j) {
    int col = nt * 64 + j * 16 + l16;
    float b = bias[col];
#pragma unroll
    for (int g = 0; g < 2; ++g)
#pragma unroll
      for (int r = 0; r < 8; ++r) {
        int row = mt * 32 + g * 16 + half * 8 + r;
        out[(size_t)row * DM + col] = acc[g * 4 + j][r] + b;
      }
  }
}

// ---------------------------------------------------------------------------
extern "C" void kernel_launch(void* const* d_in, const int* in_sizes, int n_in,
                              void* d_out, int out_size, void* d_ws, size_t ws_size,
                              hipStream_t stream) {
  const float* x  = (const float*)d_in[0];
  const float* wq = (const float*)d_in[1];
  const float* wk = (const float*)d_in[2];
  const float* wv = (const float*)d_in[3];
  const float* wo = (const float*)d_in[4];
  const float* bo = (const float*)d_in[5];
  float* out = (float*)d_out;

  char* ws = (char*)d_ws;
  size_t off = 0;
  const size_t XSZ = (size_t)SEQ * DM * sizeof(__bf16);
  const size_t WSZ = (size_t)DM * DM * sizeof(__bf16);
  __bf16* xb  = (__bf16*)(ws + off); off += XSZ;
  __bf16* wqT = (__bf16*)(ws + off); off += WSZ;
  __bf16* wkT = (__bf16*)(ws + off); off += WSZ;
  __bf16* wvT = (__bf16*)(ws + off); off += WSZ;
  __bf16* woT = (__bf16*)(ws + off); off += WSZ;
  __bf16* Qb  = (__bf16*)(ws + off); off += XSZ;
  __bf16* Kb  = (__bf16*)(ws + off); off += XSZ;
  __bf16* VTb = (__bf16*)(ws + off); off += XSZ;
  __bf16* Ctx = (__bf16*)(ws + off); off += XSZ;

  // prep: covers max(SEQ*DM, DM*DM) elements
  {
    int n = SEQ * DM;
    prep_kernel<<<(n + 255) / 256, 256, 0, stream>>>(x, wq, wk, wv, wo,
                                                     xb, wqT, wkT, wvT, woT);
  }
  // qkv: 3 matrices * 128 * 12 waves, 4 waves per 128-thread block
  {
    int waves = 3 * MT32 * NT;                 // 4608
    qkv_kernel<<<waves / 4, 128, 0, stream>>>(xb, wqT, wkT, wvT, Qb, Kb, VTb);
  }
  // attention: 12 heads * 256 query tiles
  {
    int waves = NH * MT;                       // 3072
    attn_kernel<<<waves / 4, 128, 0, stream>>>(Qb, Kb, VTb, Ctx);
  }
  // output projection: 128 * 12 waves
  {
    int waves = MT32 * NT;                     // 1536
    outproj_kernel<<<waves / 4, 128, 0, stream>>>(Ctx, woT, bo, out);
  }
}
